// GlobalCorrLayer_45835890983480
// MI455X (gfx1250) — compile-verified, matrix operand order checked by reading, not run on hardware
//
#include <hip/hip_runtime.h>
#include <hip/hip_bf16.h>

// ---------------------------------------------------------------------------
// GlobalCorrLayer pipeline for MI455X (gfx1250, wave32, WMMA, async-LDS)
// ---------------------------------------------------------------------------

typedef __attribute__((ext_vector_type(16))) _Float16 v16h;
typedef __attribute__((ext_vector_type(8)))  float    v8f;

#define BATCH 8
#define N0 8192
#define N1 1024
#define N2 512
#define N3 256
#define CFEAT 256
#define KNN 16
#define SUPPORT_TH 100.0f

// ---------------------------------------------------------------------------
// 1) Normalize features over channel dim, transpose [B,C,N] -> [B,N,C], fp16
// ---------------------------------------------------------------------------
__global__ void norm_feat_kernel(const float* __restrict__ f, _Float16* __restrict__ out) {
  int gid = blockIdx.x * blockDim.x + threadIdx.x;
  if (gid >= BATCH * N3) return;
  int b = gid / N3, n = gid % N3;
  const float* p = f + (size_t)b * CFEAT * N3 + n;   // stride N3 over channels
  float s = 0.0f;
  for (int c = 0; c < CFEAT; ++c) { float v = p[(size_t)c * N3]; s += v * v; }
  float inv = rsqrtf(s + 1e-8f);
  _Float16* o = out + ((size_t)b * N3 + n) * CFEAT;
  for (int c = 0; c < CFEAT; ++c) o[c] = (_Float16)(p[(size_t)c * N3] * inv);
}

// ---------------------------------------------------------------------------
// 2) Correlation: corr = exp(-(1 - f1n.f2n^T)/eps) * support
//    8 waves/block; each wave one 16x16 tile; K=256 fully unrolled -> 8 WMMA
// ---------------------------------------------------------------------------
__global__ __launch_bounds__(256)
void corr_wmma_kernel(const _Float16* __restrict__ f1h, const _Float16* __restrict__ f2h,
                      const float* __restrict__ p1, const float* __restrict__ p2,
                      const float* __restrict__ epsilon, float* __restrict__ corr) {
  int wave = threadIdx.x >> 5;
  int lane = threadIdx.x & 31;
  int nt = blockIdx.x;
  int mt = blockIdx.y * 8 + wave;
  int b  = blockIdx.z;
  int m  = lane & 15;
  int hi = (lane >> 4) & 1;           // selects K 8..15 / 24..31 (A,B) and M+8 (D)
  const _Float16* A  = f1h + ((size_t)b * N3 + nt * 16) * CFEAT;
  const _Float16* Bm = f2h + ((size_t)b * N3 + mt * 16) * CFEAT;
  v8f acc = {0.f, 0.f, 0.f, 0.f, 0.f, 0.f, 0.f, 0.f};
#pragma unroll
  for (int kk = 0; kk < CFEAT; kk += 32) {
    v16h a, bf;
#pragma unroll
    for (int v = 0; v < 8; ++v) {
      int kb = (v < 4 ? 2 * v : 16 + 2 * (v - 4)) + (hi ? 8 : 0) + kk;
      a[2 * v]      = A[(size_t)m * CFEAT + kb];
      a[2 * v + 1]  = A[(size_t)m * CFEAT + kb + 1];
      bf[2 * v]     = Bm[(size_t)m * CFEAT + kb];
      bf[2 * v + 1] = Bm[(size_t)m * CFEAT + kb + 1];
    }
    acc = __builtin_amdgcn_wmma_f32_16x16x32_f16(false, a, false, bf, (short)0, acc,
                                                 false, false);
  }
  float nie = -1.0f / (__expf(epsilon[0]) + 0.03f);   // hoist the divide
  const float* P1 = p1 + (size_t)b * 3 * N3;
  const float* P2 = p2 + (size_t)b * 3 * N3;
  int col = mt * 16 + m;
  float qx = P2[col], qy = P2[N3 + col], qz = P2[2 * N3 + col];
#pragma unroll
  for (int r = 0; r < 8; ++r) {
    int row = nt * 16 + r + (hi ? 8 : 0);
    float dx = P1[row] - qx, dy = P1[N3 + row] - qy, dz = P1[2 * N3 + row] - qz;
    float d2 = dx * dx + dy * dy + dz * dz;
    float Cv = 1.0f - acc[r];
    float v  = (d2 < SUPPORT_TH) ? __expf(Cv * nie) : 0.0f;
    corr[((size_t)b * N3 + row) * N3 + col] = v;
  }
}

// ---------------------------------------------------------------------------
// 3) flow0 = corr @ p2 / rowsum - p1   (one wave per row)
// ---------------------------------------------------------------------------
__global__ void flow0_kernel(const float* __restrict__ corr, const float* __restrict__ p1,
                             const float* __restrict__ p2, float* __restrict__ flow0) {
  int wid  = (blockIdx.x * blockDim.x + threadIdx.x) >> 5;
  int lane = threadIdx.x & 31;
  if (wid >= BATCH * N3) return;
  int b = wid / N3, n = wid % N3;
  const float* cr = corr + ((size_t)b * N3 + n) * N3;
  const float* P2 = p2 + (size_t)b * 3 * N3;
  float s = 0.f, sx = 0.f, sy = 0.f, sz = 0.f;
  for (int m = lane; m < N3; m += 32) {
    float c = cr[m];
    s += c; sx += c * P2[m]; sy += c * P2[N3 + m]; sz += c * P2[2 * N3 + m];
  }
  for (int off = 16; off > 0; off >>= 1) {
    s  += __shfl_down(s, off);  sx += __shfl_down(sx, off);
    sy += __shfl_down(sy, off); sz += __shfl_down(sz, off);
  }
  if (lane == 0) {
    const float* P1 = p1 + (size_t)b * 3 * N3;
    float inv = 1.0f / (s + 1e-8f);
    float* o = flow0 + ((size_t)b * N3 + n) * 3;
    o[0] = sx * inv - P1[n];
    o[1] = sy * inv - P1[N3 + n];
    o[2] = sz * inv - P1[2 * N3 + n];
  }
}

// ---------------------------------------------------------------------------
// 4) feature_prop: 3-NN inverse-distance interpolation.
//    Reference cloud staged in LDS (<=12KB of 320KB/WGP). N multiple of 256.
// ---------------------------------------------------------------------------
__global__ __launch_bounds__(256)
void fprop_kernel(const float* __restrict__ xyz1, int N,
                  const float* __restrict__ xyz2, int S,
                  const float* __restrict__ feat, int C,
                  float* __restrict__ out, int out_cn) {
  __shared__ float sh[3 * 1024];
  int bpb = N / 256;                    // blocks per batch
  int b = blockIdx.x / bpb;
  int n = (blockIdx.x % bpb) * 256 + threadIdx.x;
  const float* X2 = xyz2 + (size_t)b * 3 * S;
  for (int i = threadIdx.x; i < 3 * S; i += 256) sh[i] = X2[i];
  __syncthreads();
  const float* X1 = xyz1 + (size_t)b * 3 * N;
  float px = X1[n], py = X1[N + n], pz = X1[2 * N + n];
  float d0 = 1e30f, d1 = 1e30f, d2 = 1e30f;
  int   i0 = 0,     i1 = 0,     i2 = 0;
  for (int s = 0; s < S; ++s) {
    float dx = sh[s] - px, dy = sh[S + s] - py, dz = sh[2 * S + s] - pz;
    float d = dx * dx + dy * dy + dz * dz;
    if (d < d0)      { d2 = d1; i2 = i1; d1 = d0; i1 = i0; d0 = d; i0 = s; }
    else if (d < d1) { d2 = d1; i2 = i1; d1 = d;  i1 = s; }
    else if (d < d2) { d2 = d;  i2 = s; }
  }
  float w0 = 1.0f / (d0 + 1e-8f), w1 = 1.0f / (d1 + 1e-8f), w2 = 1.0f / (d2 + 1e-8f);
  float ws = w0 + w1 + w2;
  w0 /= ws; w1 /= ws; w2 /= ws;
  const float* F = feat + (size_t)b * S * C;
  __builtin_prefetch(F + (size_t)i0 * C, 0, 1);   // global_prefetch_b8
  __builtin_prefetch(F + (size_t)i1 * C, 0, 1);
  for (int c = 0; c < C; ++c) {
    float v = w0 * F[(size_t)i0 * C + c] + w1 * F[(size_t)i1 * C + c]
            + w2 * F[(size_t)i2 * C + c];
    if (out_cn) out[((size_t)b * C + c) * N + n] = v;
    else        out[((size_t)b * N + n) * C + c] = v;
  }
}

// ---------------------------------------------------------------------------
// 5) FPS, single wave per batch: dist[] lives in VGPRs (32 slots x 32 lanes),
//    coords in LDS, argmax via __shfl_xor butterfly -> zero barriers in the
//    serial loop (the latency bottleneck of the whole pipeline).
// ---------------------------------------------------------------------------
__global__ __launch_bounds__(32)
void fps_kernel(const float* __restrict__ xyz, int N, int npoint,
                int* __restrict__ out) {
  __shared__ float sh[3 * 1024];
  int b = blockIdx.x;
  int lane = threadIdx.x;
  const float* x = xyz + (size_t)b * 3 * N;
  for (int i = lane; i < 3 * N; i += 32) sh[i] = x[i];
  __syncthreads();
  float dist[32];
#pragma unroll
  for (int j = 0; j < 32; ++j) dist[j] = (j * 32 + lane < N) ? 1e10f : -1e30f;
  int far = 0;
  for (int it = 0; it < npoint; ++it) {
    if (lane == 0) out[b * npoint + it] = far;
    float fx = sh[far], fy = sh[N + far], fz = sh[2 * N + far];
    float best = -1e30f; int bi = 0x7fffffff;
#pragma unroll
    for (int j = 0; j < 32; ++j) {
      int i = j * 32 + lane;
      if (i < N) {
        float dx = sh[i] - fx, dy = sh[N + i] - fy, dz = sh[2 * N + i] - fz;
        float d = dx * dx + dy * dy + dz * dz;
        float nd = fminf(dist[j], d);
        dist[j] = nd;
        if (nd > best) { best = nd; bi = i; }
      }
    }
#pragma unroll
    for (int off = 16; off > 0; off >>= 1) {
      float ov = __shfl_xor(best, off);
      int   oi = __shfl_xor(bi, off);
      if (ov > best || (ov == best && oi < bi)) { best = ov; bi = oi; }
    }
    far = bi;
  }
}

// ---------------------------------------------------------------------------
// 6) kNN-16 per sampled point; point cloud staged in LDS. S multiple of 256.
// ---------------------------------------------------------------------------
__global__ __launch_bounds__(256)
void knn_kernel(const float* __restrict__ xyz, const int* __restrict__ fps,
                int N, int S, int* __restrict__ knn, float* __restrict__ nxyz) {
  __shared__ float sh[3 * 1024];
  int bpb = S / 256;
  int b = blockIdx.x / bpb;
  int s = (blockIdx.x % bpb) * 256 + threadIdx.x;
  const float* x = xyz + (size_t)b * 3 * N;
  for (int i = threadIdx.x; i < 3 * N; i += 256) sh[i] = x[i];
  __syncthreads();
  int fi = fps[b * S + s];
  float cx = sh[fi], cy = sh[N + fi], cz = sh[2 * N + fi];
  float* nz = nxyz + ((size_t)b * S + s) * 3;
  nz[0] = cx; nz[1] = cy; nz[2] = cz;
  float bd[KNN]; int bi[KNN];
#pragma unroll
  for (int k = 0; k < KNN; ++k) { bd[k] = 1e30f; bi[k] = 0; }
  for (int i = 0; i < N; ++i) {
    float dx = sh[i] - cx, dy = sh[N + i] - cy, dz = sh[2 * N + i] - cz;
    float d = dx * dx + dy * dy + dz * dz;
    if (d < bd[KNN - 1]) {
      int k = KNN - 1;
      while (k > 0 && d < bd[k - 1]) { bd[k] = bd[k - 1]; bi[k] = bi[k - 1]; --k; }
      bd[k] = d; bi[k] = i;
    }
  }
  int* o = knn + ((size_t)b * S + s) * KNN;
#pragma unroll
  for (int k = 0; k < KNN; ++k) o[k] = bi[k];
}

// ---------------------------------------------------------------------------
// 7) Group + MLP layer 0: x = [g_xyz, g_pts] @ W0   (small K, VALU) -> f32
// ---------------------------------------------------------------------------
__global__ void group_mlp0_kernel(const float* __restrict__ xyz, const float* __restrict__ nxyz,
                                  const int* __restrict__ knn, const float* __restrict__ pts,
                                  int N, int S, int Cp, const float* __restrict__ W0,
                                  int Cout, float* __restrict__ y0) {
  int gid = blockIdx.x * blockDim.x + threadIdx.x;
  if (gid >= BATCH * S * KNN) return;
  int k = gid % KNN;
  int s = (gid / KNN) % S;
  int b = gid / (KNN * S);
  int idx = knn[((size_t)b * S + s) * KNN + k];
  const float* x = xyz + (size_t)b * 3 * N;
  const float* nz = nxyz + ((size_t)b * S + s) * 3;
  float in[70];
  in[0] = x[idx] - nz[0];
  in[1] = x[N + idx] - nz[1];
  in[2] = x[2 * N + idx] - nz[2];
  const float* p = pts + ((size_t)b * N + idx) * Cp;
  for (int c = 0; c < Cp; ++c) in[3 + c] = p[c];
  int Cin = 3 + Cp;
  float* o = y0 + (size_t)gid * Cout;
  for (int j = 0; j < Cout; ++j) {
    float acc = 0.0f;
    for (int c = 0; c < Cin; ++c) acc += in[c] * W0[(size_t)c * Cout + j];
    o[j] = acc;
  }
}

// ---------------------------------------------------------------------------
// 8) Instance-norm statistics per (b, channel) over S*K elements
// ---------------------------------------------------------------------------
__global__ void in_stats_kernel(const float* __restrict__ x, int SK, int C,
                                float* __restrict__ stats) {
  int c = blockIdx.x, b = blockIdx.y;
  int t = threadIdx.x;
  const float* p = x + ((size_t)b * SK) * C + c;
  float s = 0.f, s2 = 0.f;
  for (int r = t; r < SK; r += 256) {
    float v = p[(size_t)r * C];
    s += v; s2 += v * v;
  }
  __shared__ float ss[256], sq[256];
  ss[t] = s; sq[t] = s2;
  __syncthreads();
  for (int h = 128; h > 0; h >>= 1) {
    if (t < h) { ss[t] += ss[t + h]; sq[t] += sq[t + h]; }
    __syncthreads();
  }
  if (t == 0) {
    float mu  = ss[0] / (float)SK;
    float var = sq[0] / (float)SK - mu * mu;
    stats[((size_t)b * C + c) * 2 + 0] = mu;
    stats[((size_t)b * C + c) * 2 + 1] = rsqrtf(var + 1e-5f);
  }
}

// ---------------------------------------------------------------------------
// 9) IN-affine + ReLU, f32 in -> f16 out (halves HBM traffic for next GEMM)
// ---------------------------------------------------------------------------
__global__ void in_normrelu_kernel(const float* __restrict__ x, _Float16* __restrict__ xo,
                                   int SK, int C, const float* __restrict__ stats,
                                   const float* __restrict__ g,
                                   const float* __restrict__ beta, long long total) {
  long long i = (long long)blockIdx.x * blockDim.x + threadIdx.x;
  if (i >= total) return;
  int c = (int)(i % C);
  long long row = i / C;
  int b = (int)(row / SK);
  float mu = stats[((size_t)b * C + c) * 2 + 0];
  float rs = stats[((size_t)b * C + c) * 2 + 1];
  float v = (x[i] - mu) * rs * g[c] + beta[c];
  xo[i] = (_Float16)(v > 0.0f ? v : 0.0f);
}

// ---------------------------------------------------------------------------
// 10) MLP layers 1/2: Y[R,J] = Xh[R,K] @ W[K,J], f16 A, f32 out.
//     Weights staged global->LDS with GLOBAL_LOAD_ASYNC_TO_LDS_B128
//     (ASYNCcnt path), then consumed as B fragments with on-read f16 cvt.
//     8 waves/block cover 128 rows x all J columns; K,J compile-time ->
//     fully unrolled v_wmma chains.
// ---------------------------------------------------------------------------
template <int K, int J>
__global__ __launch_bounds__(256)
void gemm_wmma_kernel(const _Float16* __restrict__ Xh, const float* __restrict__ W,
                      float* __restrict__ Y) {
  __shared__ float Wl[K * J];          // row-major [K,J] f32 staged via async DMA
  int tid = threadIdx.x;
  // async global->LDS copy, 16B per lane per issue
  for (int i4 = tid; i4 < (K * J) / 4; i4 += 256) {
    unsigned ldsoff = (unsigned)(uintptr_t)(&Wl[i4 * 4]);
    unsigned long long ga = (unsigned long long)(uintptr_t)(W + i4 * 4);
    asm volatile("global_load_async_to_lds_b128 %0, %1, off"
                 :: "v"(ldsoff), "v"(ga) : "memory");
  }
  asm volatile("s_wait_asynccnt 0x0" ::: "memory");
  __syncthreads();

  int wave = tid >> 5, lane = tid & 31;
  int m  = lane & 15;
  int hi = (lane >> 4) & 1;
  int r0 = blockIdx.x * 128 + wave * 16;
  const _Float16* xr = Xh + (size_t)(r0 + m) * K;
  v16h a[K / 32];
#pragma unroll
  for (int c = 0; c < K / 32; ++c) {
#pragma unroll
    for (int v = 0; v < 8; ++v) {
      int kb = c * 32 + (v < 4 ? 2 * v : 16 + 2 * (v - 4)) + (hi ? 8 : 0);
      a[c][2 * v]     = xr[kb];
      a[c][2 * v + 1] = xr[kb + 1];
    }
  }
#pragma unroll
  for (int jt = 0; jt < J / 16; ++jt) {
    int j = jt * 16 + m;
    v8f acc = {0.f, 0.f, 0.f, 0.f, 0.f, 0.f, 0.f, 0.f};
#pragma unroll
    for (int c = 0; c < K / 32; ++c) {
      v16h bf;
#pragma unroll
      for (int v = 0; v < 8; ++v) {
        int kb = c * 32 + (v < 4 ? 2 * v : 16 + 2 * (v - 4)) + (hi ? 8 : 0);
        bf[2 * v]     = (_Float16)Wl[kb * J + j];
        bf[2 * v + 1] = (_Float16)Wl[(kb + 1) * J + j];
      }
      acc = __builtin_amdgcn_wmma_f32_16x16x32_f16(false, a[c], false, bf, (short)0, acc,
                                                   false, false);
    }
#pragma unroll
    for (int r = 0; r < 8; ++r) {
      int row = r0 + r + (hi ? 8 : 0);
      Y[(size_t)row * J + j] = acc[r];
    }
  }
}

// ---------------------------------------------------------------------------
// 11) Max over the K=16 neighbor axis (f16 in) -> [B,S,C] f32
// ---------------------------------------------------------------------------
__global__ void maxpool_kernel(const _Float16* __restrict__ x, int S, int C,
                               float* __restrict__ out) {
  int gid = blockIdx.x * blockDim.x + threadIdx.x;
  if (gid >= BATCH * S * C) return;
  int c = gid % C;
  int s = (gid / C) % S;
  int b = gid / (C * S);
  const _Float16* p = x + ((size_t)(b * S + s) * KNN) * C + c;
  float m = -1e30f;
#pragma unroll
  for (int k = 0; k < KNN; ++k) m = fmaxf(m, (float)p[(size_t)k * C]);
  out[((size_t)b * S + s) * C + c] = m;
}

// ---------------------------------------------------------------------------
// Host side
// ---------------------------------------------------------------------------
static inline size_t align256(size_t v) { return (v + 255) & ~(size_t)255; }

extern "C" void kernel_launch(void* const* d_in, const int* in_sizes, int n_in,
                              void* d_out, int out_size, void* d_ws, size_t ws_size,
                              hipStream_t stream) {
  (void)in_sizes; (void)n_in; (void)out_size; (void)ws_size;
  const float* pc1_l0 = (const float*)d_in[0];
  const float* pc1_l1 = (const float*)d_in[1];
  const float* pc1_l2 = (const float*)d_in[2];
  const float* pc1_l3 = (const float*)d_in[3];
  const float* pc2_l3 = (const float*)d_in[4];
  const float* feats1 = (const float*)d_in[5];
  const float* feats2 = (const float*)d_in[6];
  const float* epsv   = (const float*)d_in[7];
  const float* sa1_w0 = (const float*)d_in[8];
  const float* sa1_w1 = (const float*)d_in[9];
  const float* sa1_w2 = (const float*)d_in[10];
  const float* sa1_g0 = (const float*)d_in[11];
  const float* sa1_b0 = (const float*)d_in[12];
  const float* sa1_g1 = (const float*)d_in[13];
  const float* sa1_b1 = (const float*)d_in[14];
  const float* sa1_g2 = (const float*)d_in[15];
  const float* sa1_b2 = (const float*)d_in[16];
  const float* sa2_w0 = (const float*)d_in[17];
  const float* sa2_w1 = (const float*)d_in[18];
  const float* sa2_w2 = (const float*)d_in[19];
  const float* sa2_g0 = (const float*)d_in[20];
  const float* sa2_b0 = (const float*)d_in[21];
  const float* sa2_g1 = (const float*)d_in[22];
  const float* sa2_b1 = (const float*)d_in[23];
  const float* sa2_g2 = (const float*)d_in[24];
  const float* sa2_b2 = (const float*)d_in[25];
  float* out = (float*)d_out;

  // deterministic bump allocator over workspace
  char* ws = (char*)d_ws;
  size_t off = 0;
  auto alloc = [&](size_t bytes) { char* p = ws + off; off = align256(off + bytes); return (void*)p; };

  const int R1 = BATCH * N2 * KNN;   // 65536
  const int R2 = BATCH * N1 * KNN;   // 131072

  _Float16* f1h  = (_Float16*)alloc((size_t)BATCH * N3 * CFEAT * 2);
  _Float16* f2h  = (_Float16*)alloc((size_t)BATCH * N3 * CFEAT * 2);
  float* corr    = (float*)alloc((size_t)BATCH * N3 * N3 * 4);
  float* flow0   = (float*)alloc((size_t)BATCH * N3 * 3 * 4);
  float* flow_us = (float*)alloc((size_t)BATCH * N2 * 3 * 4);
  int*   fps1    = (int*)alloc((size_t)BATCH * N2 * 4);
  int*   knn1    = (int*)alloc((size_t)BATCH * N2 * KNN * 4);
  float* nxyz1   = (float*)alloc((size_t)BATCH * N2 * 3 * 4);
  int*   fps2    = (int*)alloc((size_t)BATCH * N1 * 4);
  int*   knn2    = (int*)alloc((size_t)BATCH * N1 * KNN * 4);
  float* nxyz2   = (float*)alloc((size_t)BATCH * N1 * 3 * 4);
  float* stats   = (float*)alloc((size_t)BATCH * 128 * 2 * 4);
  float* cf2     = (float*)alloc((size_t)BATCH * N2 * 64 * 4);
  float* cf1     = (float*)alloc((size_t)BATCH * N1 * 64 * 4);
  float* cf1b    = (float*)alloc((size_t)BATCH * N1 * 128 * 4);
  float*    poolA  = (float*)alloc((size_t)R2 * 64 * 4);      // layer0/1 f32 outputs
  float*    poolB  = (float*)alloc((size_t)R2 * 64 * 4);
  float*    poolC  = (float*)alloc((size_t)R2 * 128 * 4);     // layer2 f32 outputs
  _Float16* poolAh = (_Float16*)alloc((size_t)R2 * 64 * 2);   // f16 activations
  _Float16* poolBh = (_Float16*)alloc((size_t)R2 * 64 * 2);
  _Float16* poolCh = (_Float16*)alloc((size_t)R2 * 128 * 2);

  // ---- Stage 1: global correlation -> flow0 ----
  norm_feat_kernel<<<(BATCH * N3 + 255) / 256, 256, 0, stream>>>(feats1, f1h);
  norm_feat_kernel<<<(BATCH * N3 + 255) / 256, 256, 0, stream>>>(feats2, f2h);
  corr_wmma_kernel<<<dim3(N3 / 16, (N3 / 16) / 8, BATCH), 256, 0, stream>>>(
      f1h, f2h, pc1_l3, pc2_l3, epsv, corr);
  flow0_kernel<<<(BATCH * N3 * 32 + 255) / 256, 256, 0, stream>>>(corr, pc1_l3, pc2_l3, flow0);

  // ---- flow0 upsample 256 -> 512 ----
  fprop_kernel<<<BATCH * (N2 / 256), 256, 0, stream>>>(
      pc1_l2, N2, pc1_l3, N3, flow0, 3, flow_us, 0);

  // ---- SA1: N=512, npoint=512, mlp 6->32->32->64 ----
  fps_kernel<<<BATCH, 32, 0, stream>>>(pc1_l2, N2, N2, fps1);
  knn_kernel<<<BATCH * (N2 / 256), 256, 0, stream>>>(pc1_l2, fps1, N2, N2, knn1, nxyz1);
  group_mlp0_kernel<<<(R1 + 255) / 256, 256, 0, stream>>>(
      pc1_l2, nxyz1, knn1, flow_us, N2, N2, 3, sa1_w0, 32, poolA);
  {
    const int SK = N2 * KNN;
    long long t0 = (long long)R1 * 32;
    in_stats_kernel<<<dim3(32, BATCH), 256, 0, stream>>>(poolA, SK, 32, stats);
    in_normrelu_kernel<<<(unsigned)((t0 + 255) / 256), 256, 0, stream>>>(
        poolA, poolAh, SK, 32, stats, sa1_g0, sa1_b0, t0);
    gemm_wmma_kernel<32, 32><<<R1 / 128, 256, 0, stream>>>(poolAh, sa1_w1, poolB);
    in_stats_kernel<<<dim3(32, BATCH), 256, 0, stream>>>(poolB, SK, 32, stats);
    in_normrelu_kernel<<<(unsigned)((t0 + 255) / 256), 256, 0, stream>>>(
        poolB, poolBh, SK, 32, stats, sa1_g1, sa1_b1, t0);
    gemm_wmma_kernel<32, 64><<<R1 / 128, 256, 0, stream>>>(poolBh, sa1_w2, poolC);
    long long t2 = (long long)R1 * 64;
    in_stats_kernel<<<dim3(64, BATCH), 256, 0, stream>>>(poolC, SK, 64, stats);
    in_normrelu_kernel<<<(unsigned)((t2 + 255) / 256), 256, 0, stream>>>(
        poolC, poolCh, SK, 64, stats, sa1_g2, sa1_b2, t2);
    maxpool_kernel<<<(BATCH * N2 * 64 + 255) / 256, 256, 0, stream>>>(poolCh, N2, 64, cf2);
  }

  // ---- propagate 512 -> 1024 ----
  fprop_kernel<<<BATCH * (N1 / 256), 256, 0, stream>>>(
      pc1_l1, N1, pc1_l2, N2, cf2, 64, cf1, 0);

  // ---- SA2: N=1024, npoint=1024, mlp 67->64->64->128 ----
  fps_kernel<<<BATCH, 32, 0, stream>>>(pc1_l1, N1, N1, fps2);
  knn_kernel<<<BATCH * (N1 / 256), 256, 0, stream>>>(pc1_l1, fps2, N1, N1, knn2, nxyz2);
  group_mlp0_kernel<<<(R2 + 255) / 256, 256, 0, stream>>>(
      pc1_l1, nxyz2, knn2, cf1, N1, N1, 64, sa2_w0, 64, poolA);
  {
    const int SK = N1 * KNN;
    long long t0 = (long long)R2 * 64;
    in_stats_kernel<<<dim3(64, BATCH), 256, 0, stream>>>(poolA, SK, 64, stats);
    in_normrelu_kernel<<<(unsigned)((t0 + 255) / 256), 256, 0, stream>>>(
        poolA, poolAh, SK, 64, stats, sa2_g0, sa2_b0, t0);
    gemm_wmma_kernel<64, 64><<<R2 / 128, 256, 0, stream>>>(poolAh, sa2_w1, poolB);
    in_stats_kernel<<<dim3(64, BATCH), 256, 0, stream>>>(poolB, SK, 64, stats);
    in_normrelu_kernel<<<(unsigned)((t0 + 255) / 256), 256, 0, stream>>>(
        poolB, poolBh, SK, 64, stats, sa2_g1, sa2_b1, t0);
    gemm_wmma_kernel<64, 128><<<R2 / 128, 256, 0, stream>>>(poolBh, sa2_w2, poolC);
    long long t2 = (long long)R2 * 128;
    in_stats_kernel<<<dim3(128, BATCH), 256, 0, stream>>>(poolC, SK, 128, stats);
    in_normrelu_kernel<<<(unsigned)((t2 + 255) / 256), 256, 0, stream>>>(
        poolC, poolCh, SK, 128, stats, sa2_g2, sa2_b2, t2);
    maxpool_kernel<<<(BATCH * N1 * 128 + 255) / 256, 256, 0, stream>>>(poolCh, N1, 128, cf1b);
  }

  // ---- final propagate 1024 -> 8192, write [B,128,8192] ----
  fprop_kernel<<<BATCH * (N0 / 256), 256, 0, stream>>>(
      pc1_l0, N0, pc1_l1, N1, cf1b, 128, out, 1);
}
